// BiESN_90228672954699
// MI455X (gfx1250) — matrix operand order repaired and checked by reading, not compiled, not run on hardware
//
#include <hip/hip_runtime.h>
#include <hip/hip_bf16.h>

// ---------------------------------------------------------------------------
// Bi-directional Echo State Network for MI455X (gfx1250, wave32, WMMA).
//   B=32, T=2048, D=512, U=512, leaky=0.9
// Phase 0: convert kernel_f/b and rec_f/b to bf16 (workspace).
// Phase 1: xin[dir][b][t][u] = sum_d x[b][t][d]*kernel[d][u]   (bf16 WMMA GEMM)
// Phase 2: sequential scan, 4 workgroups (2 dirs x 2 batch halves of M=16),
//          32 waves per workgroup, one 16-wide N-tile per wave.
//          rec B-fragments resident in VGPRs; h state bf16 in ping-pong LDS
//          (A fragments), fp32 state kept in registers for the leaky blend.
//          xin tile for step s+1 prefetched during step s's WMMA chain;
//          one workgroup barrier per step.
// ---------------------------------------------------------------------------

#define B_   32
#define T_   2048
#define D_   512
#define U_   512
#define KT_  16           // number of K-tiles of 32 covering K=512
#define LDSROW 520        // bf16 elements per h row (512 + 8 pad, 16B aligned)

typedef __attribute__((ext_vector_type(16))) __bf16 v16bf;
typedef __attribute__((ext_vector_type(8)))  __bf16 v8bf;
typedef __attribute__((ext_vector_type(8)))  float  v8f;

// ---------------------------------------------------------------- Phase 0 --
__global__ void f32_to_bf16_kernel(const float* __restrict__ s,
                                   __bf16* __restrict__ d, int n) {
  int i = blockIdx.x * blockDim.x + threadIdx.x;
  if (i < n) d[i] = (__bf16)s[i];
}

// ---------------------------------------------------------------- Phase 1 --
// grid = (32, 64), block = 256 (8 waves). blockIdx.y = N-tile (0..31 fwd kernel,
// 32..63 bwd kernel). Each wave amortizes its resident B fragments over 16
// M-tiles of the flattened [B*T] row dimension.
__global__ __launch_bounds__(256) void esn_xin_kernel(
    const float* __restrict__ x,     // [B*T][D] fp32
    const __bf16* __restrict__ kf,   // [D][U] bf16
    const __bf16* __restrict__ kb,   // [D][U] bf16
    __bf16* __restrict__ xin) {      // [2][B*T][U] bf16
  const int wave = threadIdx.x >> 5;
  const int lane = threadIdx.x & 31;
  const int ntile = blockIdx.y;
  const int dir = ntile >> 5;
  const int n0 = (ntile & 31) * 16;
  const int nn = n0 + (lane & 15);
  const int hi = lane >> 4;               // lane half: 0 or 1
  const __bf16* __restrict__ kmat = dir ? kb : kf;

  // Resident B fragments: Bf[kt] element j = kernel[kt*32 + hi*8 + (j<8 ? j : j+8)][nn]
  v16bf Bf[KT_];
  for (int kt = 0; kt < KT_; ++kt) {
    const int kb0 = kt * 32 + hi * 8;
    v16bf f;
#pragma unroll
    for (int j = 0; j < 16; ++j)
      f[j] = kmat[(size_t)(kb0 + (j < 8 ? j : j + 8)) * U_ + nn];
    Bf[kt] = f;
  }

  __bf16* __restrict__ xo = xin + (size_t)dir * B_ * T_ * U_;
  const int mt_base = (blockIdx.x * 8 + wave) * 16;   // 16 M-tiles per wave
  const int c1 = hi ? 8 : 0;

  for (int mi = 0; mi < 16; ++mi) {
    const int row0 = (mt_base + mi) * 16;
    const int arow = row0 + (lane & 15);
    v8f acc;
#pragma unroll
    for (int i = 0; i < 8; ++i) acc[i] = 0.0f;

#pragma unroll
    for (int kt = 0; kt < KT_; ++kt) {
      const int base = kt * 32 + c1;
      const float* ap = x + (size_t)arow * D_ + base;
      float fa[16];
      *(float4*)&fa[0]  = *(const float4*)(ap + 0);
      *(float4*)&fa[4]  = *(const float4*)(ap + 4);
      *(float4*)&fa[8]  = *(const float4*)(ap + 16);
      *(float4*)&fa[12] = *(const float4*)(ap + 20);
      v16bf a;
#pragma unroll
      for (int j = 0; j < 16; ++j) a[j] = (__bf16)fa[j];
      acc = __builtin_amdgcn_wmma_f32_16x16x32_bf16(
          false, a, false, Bf[kt], (short)0, acc, false, false);
    }
#pragma unroll
    for (int i = 0; i < 8; ++i) {
      const int row = row0 + hi * 8 + i;      // D-layout: m = i + 8*laneHalf
      xo[(size_t)row * U_ + nn] = (__bf16)acc[i];
    }
  }
}

// ---------------------------------------------------------------- Phase 2 --
// grid = 4 blocks x 1024 threads. block -> (dir, batch half of 16 rows).
// Wave w owns N columns [16w, 16w+16). rec B fragments resident in VGPRs.
// Ping-pong h in LDS: step s reads hbuf[s&1], writes hbuf[1-(s&1)];
// single barrier per step covers both RAW and WAR.
__global__ __launch_bounds__(1024, 1) void esn_scan_kernel(
    const __bf16* __restrict__ xin,   // [2][B*T][U] bf16
    const __bf16* __restrict__ recf,  // [U][U] bf16
    const __bf16* __restrict__ recb,  // [U][U] bf16
    const float* __restrict__ biasf,
    const float* __restrict__ biasb,
    float* __restrict__ out) {        // [B][T][2U] fp32
  __shared__ __bf16 hbuf[2][16 * LDSROW];

  const int wave = threadIdx.x >> 5;
  const int lane = threadIdx.x & 31;
  const int dir = blockIdx.x >> 1;            // 0 = fwd, 1 = bwd
  const int b0 = (blockIdx.x & 1) * 16;       // batch-row base
  const int n0 = wave * 16;
  const int nn = n0 + (lane & 15);
  const int hi = lane >> 4;
  const int mrow = hi * 8;
  const int c1 = hi ? 8 : 0;

  const __bf16* __restrict__ rec = dir ? recb : recf;
  const float bias_n = (dir ? biasb : biasf)[nn];

  // Resident rec B fragments (16 KB per wave -> 128 VGPRs).
  v16bf Bf[KT_];
  for (int kt = 0; kt < KT_; ++kt) {
    const int kb0 = kt * 32 + hi * 8;
    v16bf f;
#pragma unroll
    for (int j = 0; j < 16; ++j)
      f[j] = rec[(size_t)(kb0 + (j < 8 ? j : j + 8)) * U_ + nn];
    Bf[kt] = f;
  }

  // h0 = 0 (only the buffer read at s=0 needs zeroing)
  for (int idx = threadIdx.x; idx < 16 * LDSROW; idx += blockDim.x)
    hbuf[0][idx] = (__bf16)0.0f;

  float h[8];
#pragma unroll
  for (int i = 0; i < 8; ++i) h[i] = 0.0f;

  const __bf16* __restrict__ xin_d = xin + (size_t)dir * B_ * T_ * U_;
  const int arow = (lane & 15) * LDSROW;

  // Per-lane constant base offsets; per-step offsets are t*U (xin) / t*2U (out).
  const size_t mstride_x = (size_t)T_ * U_;        // 1 << 20 elements
  const size_t mstride_o = (size_t)T_ * 2 * U_;    // 1 << 21 elements
  const size_t xbase = (size_t)(b0 + mrow) * T_ * U_ + nn;
  const size_t obase = (size_t)(b0 + mrow) * T_ * 2 * U_ + (size_t)dir * U_ + nn;

  // Prefetch xin tile for step s=0.
  __bf16 xn[8];
  {
    const int t0 = dir ? (T_ - 1) : 0;
    const size_t toff = (size_t)t0 * U_;
#pragma unroll
    for (int i = 0; i < 8; ++i)
      xn[i] = xin_d[xbase + (size_t)i * mstride_x + toff];
  }
  __syncthreads();   // h0 zero-fill visible

  for (int s = 0; s < T_; ++s) {
    const int t = dir ? (T_ - 1 - s) : s;
    const __bf16* __restrict__ hrd = hbuf[s & 1];
    __bf16* __restrict__ hwr = hbuf[1 - (s & 1)];

    // C init from the tile prefetched during the previous step.
    v8f acc;
#pragma unroll
    for (int i = 0; i < 8; ++i) acc[i] = bias_n + (float)xn[i];

    // Prefetch xin tile for step s+1 (completes under the WMMA chain).
    {
      const int sn = (s + 1 < T_) ? (s + 1) : s;
      const int tn = dir ? (T_ - 1 - sn) : sn;
      const size_t toff = (size_t)tn * U_;
#pragma unroll
      for (int i = 0; i < 8; ++i)
        xn[i] = xin_d[xbase + (size_t)i * mstride_x + toff];
    }

    // h @ rec : A fragments from LDS, 16x v_wmma_f32_16x16x32_bf16.
#pragma unroll
    for (int kt = 0; kt < KT_; ++kt) {
      const int base = kt * 32 + c1;
      union { v16bf v; v8bf p[2]; } u;
      u.p[0] = *(const v8bf*)&hrd[arow + base];        // K = base .. base+7
      u.p[1] = *(const v8bf*)&hrd[arow + base + 16];   // K = base+16 .. +23
      acc = __builtin_amdgcn_wmma_f32_16x16x32_bf16(
          false, u.v, false, Bf[kt], (short)0, acc, false, false);
    }

    // h_new = 0.1*h + 0.9*tanh(acc) ; tanh via v_exp_f32 + v_rcp_f32.
    const size_t toff_o = (size_t)t * 2 * U_;
#pragma unroll
    for (int i = 0; i < 8; ++i) {
      const float xv = acc[i];
      const float e2x = __builtin_amdgcn_exp2f(xv * 2.8853900817779268f);
      const float th = 1.0f - 2.0f * __builtin_amdgcn_rcpf(e2x + 1.0f);
      const float hn = 0.1f * h[i] + 0.9f * th;
      h[i] = hn;
      hwr[(mrow + i) * LDSROW + nn] = (__bf16)hn;
      out[obase + (size_t)i * mstride_o + toff_o] = hn;
    }
    __syncthreads();   // h(t) visible; also protects WAR on hbuf[s&1]
  }
}

// ------------------------------------------------------------------ launch --
extern "C" void kernel_launch(void* const* d_in, const int* in_sizes, int n_in,
                              void* d_out, int out_size, void* d_ws,
                              size_t ws_size, hipStream_t stream) {
  const float* x        = (const float*)d_in[0];
  const float* kernel_f = (const float*)d_in[1];
  const float* rec_f    = (const float*)d_in[2];
  const float* bias_f   = (const float*)d_in[3];
  const float* kernel_b = (const float*)d_in[4];
  const float* rec_b    = (const float*)d_in[5];
  const float* bias_b   = (const float*)d_in[6];
  float* out = (float*)d_out;

  char* ws = (char*)d_ws;
  const size_t xin_elems = (size_t)2 * B_ * T_ * U_;
  __bf16* xin = (__bf16*)ws;
  __bf16* kf  = (__bf16*)(ws + xin_elems * 2);
  __bf16* kb  = kf + (size_t)D_ * U_;
  __bf16* rf  = kb + (size_t)D_ * U_;
  __bf16* rb  = rf + (size_t)U_ * U_;

  const int nmat = D_ * U_;   // 262144
  f32_to_bf16_kernel<<<(nmat + 255) / 256, 256, 0, stream>>>(kernel_f, kf, nmat);
  f32_to_bf16_kernel<<<(nmat + 255) / 256, 256, 0, stream>>>(kernel_b, kb, nmat);
  f32_to_bf16_kernel<<<(nmat + 255) / 256, 256, 0, stream>>>(rec_f, rf, nmat);
  f32_to_bf16_kernel<<<(nmat + 255) / 256, 256, 0, stream>>>(rec_b, rb, nmat);

  esn_xin_kernel<<<dim3(32, 64), 256, 0, stream>>>(x, kf, kb, xin);
  esn_scan_kernel<<<4, 1024, 0, stream>>>(xin, rf, rb, bias_f, bias_b, out);
}
